// OneToOneNNParallel_2035814498782
// MI455X (gfx1250) — matrix operand order repaired and checked by reading, not compiled, not run on hardware
//
#include <hip/hip_runtime.h>
#include <hip/hip_bf16.h>

// N=512 batch, D=64, H=128 hidden.
// out[n,i,j] = relu( relu(x[n,i]*w1[i,j,:]+b1[i,j,:]) @ w2[i,j,:,:] + b2[i,j,:] ) @ w3[i,j,:,0] + b3[i,j,0]

typedef __attribute__((ext_vector_type(16))) __bf16 v16bf;
typedef __attribute__((ext_vector_type(8)))  float  v8f;
typedef __attribute__((ext_vector_type(4)))  float  v4f;

#define NB   512
#define DD   64
#define HH   128

__global__ __launch_bounds__(256)
void one2one_mlp_wmma_kernel(const float* __restrict__ x,
                             const float* __restrict__ w1,
                             const float* __restrict__ b1,
                             const float* __restrict__ w2,
                             const float* __restrict__ b2,
                             const float* __restrict__ w3,
                             const float* __restrict__ b3,
                             float* __restrict__ out) {
    // One workgroup per (i,j) pair.
    const int i = blockIdx.x;
    const int j = blockIdx.y;
    const int t = threadIdx.x;                 // 0..255
    const int lane = t & 31;
    const int wave = t >> 5;                   // 0..7
    const int col  = lane & 15;                // N within tile / M within tile
    const int hi   = lane >> 4;                // 0 or 1 (half-wave)

    // LDS: B matrix (w2 tile) pre-swizzled into WMMA B-fragment layout (bf16),
    // plus staged per-(i,j) vectors.
    __shared__ __align__(16) __bf16 sB[4 * 8 * 512];   // 32 KB: [kc][nt][lane*16+e]
    __shared__ __align__(16) float sX[NB];             // x[:, i]    (2 KB)
    __shared__ __align__(16) float sW1[HH];
    __shared__ __align__(16) float sB1[HH];
    __shared__ __align__(16) float sB2[HH];
    __shared__ __align__(16) float sW3[HH];

    const long pairOff = (long)(i * DD + j);

    // ---------------- Phase 0: stage operands into LDS ----------------
    if (t < HH) {
        const long cb = pairOff * HH + t;
        sW1[t] = w1[cb];
        sB1[t] = b1[cb];
        sB2[t] = b2[cb];
        sW3[t] = w3[cb];   // w3 shape [D,D,H,1] -> same flat layout
    }
    // x column (strided gather; x is tiny and L2-resident across the 64 j-blocks)
    for (int n = t; n < NB; n += 256) sX[n] = x[(long)n * DD + i];

    // w2 tile: 128x128 f32, streamed once -> non-temporal loads, bf16 convert,
    // scatter into B-fragment-swizzled LDS.
    {
        const v4f* __restrict__ w2v =
            (const v4f*)(w2 + (pairOff << 14));         // 16384 f32 per tile
        #pragma unroll 4
        for (int it = 0; it < 16; ++it) {
            const int idx4 = it * 256 + t;              // 0..4095 float4
            v4f v = __builtin_nontemporal_load(&w2v[idx4]);
            const int base = idx4 << 2;                 // element index
            const int k    = base >> 7;                 // 0..127 (K dim)
            const int l    = base & 127;                // 0..124 step 4 (N dim)
            const int kc   = k >> 5;
            const int kmod = k & 31;
            const int e    = kmod & 15;                 // element slot in lane
            const int laneHi = kmod & 16;               // 0 or 16
            const int nt   = l >> 4;                    // same for all 4 elems
            const int fragBase = ((kc * 8 + nt) << 9);  // *512 bf16
            #pragma unroll
            for (int c = 0; c < 4; ++c) {
                const int ncol = (l + c) & 15;
                sB[fragBase + ((ncol + laneHi) << 4) + e] = (__bf16)v[c];
            }
        }
    }
    __syncthreads();

    const float b3v = b3[pairOff];

    // Unused (hi*8); the A-fragment's K-base is 16*hi within each kc chunk (see rb below).
    const int kbaseA = hi << 3;

    // Hoist the epilogue constants for this lane's column set (one per nt).
    float b2v[8], w3v[8];
    #pragma unroll
    for (int nt = 0; nt < 8; ++nt) {
        b2v[nt] = sB2[nt * 16 + col];
        w3v[nt] = sW3[nt * 16 + col];
    }

    const v4f* __restrict__ sW1v = (const v4f*)sW1;
    const v4f* __restrict__ sB1v = (const v4f*)sB1;

    // ---------------- Phase 1: per-wave GEMM + fused epilogue ----------------
    // Wave owns M-tiles mt = wave*4 .. wave*4+3 (M = 512 = 32 tiles of 16).
    for (int q = 0; q < 4; ++q) {
        const int mt = wave * 4 + q;
        const float xv = sX[mt * 16 + col];   // row value for this lane's M

        // All 8 N-tile accumulators stay live across the K loop (64 VGPRs).
        v8f acc[8];
        #pragma unroll
        for (int nt = 0; nt < 8; ++nt) acc[nt] = (v8f){};

        // K loop kept rolled: one A-fragment live at a time.
        #pragma unroll 1
        for (int kc = 0; kc < 4; ++kc) {
            // Lane's 16 (w1,b1) values: one contiguous run of 16 f32
            // (K = kc*32 + hi*16 + e) -> 4x ds_load_b128 per array.
            const int rb = kc * 8 + (hi << 2);          // v4f index of run0
            float w1f[16], b1f[16];
            *(v4f*)&w1f[0]  = sW1v[rb];
            *(v4f*)&w1f[4]  = sW1v[rb + 1];
            *(v4f*)&w1f[8]  = sW1v[rb + 2];
            *(v4f*)&w1f[12] = sW1v[rb + 3];
            *(v4f*)&b1f[0]  = sB1v[rb];
            *(v4f*)&b1f[4]  = sB1v[rb + 1];
            *(v4f*)&b1f[8]  = sB1v[rb + 2];
            *(v4f*)&b1f[12] = sB1v[rb + 3];

            // Build the A-fragment from the rank-1 layer 1 (fma+relu+bf16).
            v16bf afrag;
            #pragma unroll
            for (int e = 0; e < 16; ++e) {
                const float h = fmaxf(fmaf(xv, w1f[e], b1f[e]), 0.0f);
                afrag[e] = (__bf16)h;
            }

            // 8 WMMAs sharing this A-fragment, each into its own accumulator.
            #pragma unroll
            for (int nt = 0; nt < 8; ++nt) {
                const v16bf bfrag =
                    *(const v16bf*)(&sB[((kc * 8 + nt) << 9) + (lane << 4)]);
                acc[nt] = __builtin_amdgcn_wmma_f32_16x16x32_bf16(
                            false, afrag, false, bfrag,
                            (short)0, acc[nt], false, false);
            }
        }

        // Fused layer-2 bias + relu + layer-3 weight; per-lane partial sums.
        float pz[8] = {0.f, 0.f, 0.f, 0.f, 0.f, 0.f, 0.f, 0.f};
        #pragma unroll
        for (int nt = 0; nt < 8; ++nt) {
            #pragma unroll
            for (int r = 0; r < 8; ++r)
                pz[r] += fmaxf(acc[nt][r] + b2v[nt], 0.0f) * w3v[nt];
        }

        // Reduce pz across the 16 lanes holding the row (butterfly within half-wave).
        #pragma unroll
        for (int m = 1; m < 16; m <<= 1) {
            #pragma unroll
            for (int r = 0; r < 8; ++r)
                pz[r] += __shfl_xor(pz[r], m, 32);
        }

        // Lanes 0 and 16 now hold row sums for M = r and M = r+8 respectively.
        if (col == 0) {
            const int nbase = mt * 16 + hi * 8;
            #pragma unroll
            for (int r = 0; r < 8; ++r)
                out[((long)(nbase + r) * DD + i) * DD + j] = pz[r] + b3v;
        }
    }
}

extern "C" void kernel_launch(void* const* d_in, const int* in_sizes, int n_in,
                              void* d_out, int out_size, void* d_ws, size_t ws_size,
                              hipStream_t stream) {
    const float* x  = (const float*)d_in[0];
    const float* w1 = (const float*)d_in[1];
    const float* b1 = (const float*)d_in[2];
    const float* w2 = (const float*)d_in[3];
    const float* b2 = (const float*)d_in[4];
    const float* w3 = (const float*)d_in[5];
    const float* b3 = (const float*)d_in[6];
    float* out = (float*)d_out;

    dim3 grid(DD, DD, 1);   // one workgroup per (i,j)
    dim3 block(256, 1, 1);  // 8 waves (wave32)
    one2one_mlp_wmma_kernel<<<grid, block, 0, stream>>>(x, w1, b1, w2, b2, w3, b3, out);
}
